// LSTM_34857954574503
// MI455X (gfx1250) — matrix operand order repaired and checked by reading, not compiled, not run on hardware
//
#include <hip/hip_runtime.h>
#include <hip/hip_bf16.h>

// Problem constants (from reference)
#define LNUM 2
#define BNUM 256
#define TNUM 512
#define SNUM 8
#define HNUM 256
#define ZNUM 264
#define DINN 248
#define KDIM 512          // S + DIN + H
#define GDIM 1024         // 4*H
#define ROWS 512          // L*B
#define NWG  64
#define NTHR 256
#define LDS_STRIDE 520    // 512 + 8 ushort pad: row stride 1040B -> 4-bank rotation,
                          // so 16 lanes x 16B fragment loads hit 64 distinct banks

// Output flat offsets (outputs, z_final, c_z_final, coefficients)
#define OFF_ZF   ((size_t)34603008)   // B*T*Z
#define OFF_CZF  ((size_t)34738176)   // + L*B*Z
#define OFF_COEF ((size_t)34873344)   // + L*B*Z

typedef __attribute__((ext_vector_type(16))) __bf16 v16bf;
typedef __attribute__((ext_vector_type(8)))  float  v8f;

union FragBF { int4 q[2]; v16bf v; };

__device__ inline unsigned short f2bfu(float f) {
  unsigned u = __float_as_uint(f);
  unsigned r = u + 0x7FFFu + ((u >> 16) & 1u);   // round-to-nearest-even
  return (unsigned short)(r >> 16);
}

__device__ inline float sigm(float x) { return 1.0f / (1.0f + __expf(-x)); }

// Grid-wide split barrier: atomic arrive + generation spin (s_sleep while waiting).
__device__ inline void grid_barrier(unsigned* bar) {
  __threadfence();
  __syncthreads();
  if (threadIdx.x == 0) {
    volatile unsigned* vb = (volatile unsigned*)bar;
    unsigned gen = vb[1];
    __threadfence();
    unsigned arrived = atomicAdd(&bar[0], 1u);
    if (arrived == NWG - 1) {
      vb[0] = 0u;
      __threadfence();
      atomicAdd(&bar[1], 1u);
    } else {
      while (vb[1] == gen) { __builtin_amdgcn_s_sleep(2); }
      __threadfence();
    }
  }
  __syncthreads();
}

__global__ void bar_init_kernel(unsigned* bar) { bar[0] = 0u; bar[1] = 0u; }

__global__ __launch_bounds__(NTHR, 1) void ltc_lstm_persistent(
    const float* __restrict__ u,    // (B,T,DIN)
    const float* __restrict__ z0,   // (L,B,Z)
    const float* __restrict__ cz0,  // (L,B,Z)
    const float* __restrict__ tau,  // (B,T,1)
    const float* __restrict__ Amat, // (S,S)
    const float* __restrict__ W,    // (L,4H,K)
    const float* __restrict__ bias, // (L,4H)
    const float* __restrict__ aW,   // (S,H)
    const float* __restrict__ ab,   // (S)
    const float* __restrict__ xWm,  // (S,H)
    const float* __restrict__ xbv,  // (S)
    float* __restrict__ out,
    float* __restrict__ x32,            // (ROWS, S)
    float* __restrict__ c32,            // (ROWS, H)  workgroup-private slices
    float* __restrict__ h32,            // (ROWS, H)
    unsigned short* __restrict__ hbf,   // (ROWS, H) bf16
    unsigned* __restrict__ bar) {
  const int g    = blockIdx.x;
  const int tid  = threadIdx.x;
  const int gt   = g * NTHR + tid;       // 0..16383
  const int wave = tid >> 5;
  const int lane = tid & 31;

  // WG ownership: layer l, 64-row group rtg, 32-h-col group cg
  const int l   = g >> 5;        // 0..1
  const int rtg = (g >> 3) & 3;  // 0..3
  const int cg  = g & 7;         // 0..7
  const int r0  = l * 256 + rtg * 64;

  // LDS-resident operands (persist across all 512 steps for sB/sAW/sXW)
  __shared__ __align__(16) unsigned short sA[64 * LDS_STRIDE];   // 66.5 KB activations
  __shared__ __align__(16) unsigned short sB[128 * LDS_STRIDE];  // 133 KB: this WG's W slice
  __shared__ float sAW[SNUM * HNUM];                             // 8 KB
  __shared__ float sXW[SNUM * HNUM];                             // 8 KB

  // ---------- init: global state from z0/cz0 ----------
  for (int i = gt; i < ROWS * HNUM; i += NWG * NTHR) {
    int r = i >> 8, j = i & 255;
    float h0 = z0[r * ZNUM + SNUM + j];
    h32[i] = h0;
    hbf[i] = f2bfu(h0);
    c32[i] = cz0[r * ZNUM + SNUM + j];
  }
  for (int i = gt; i < ROWS * SNUM; i += NWG * NTHR) {
    int r = i >> 3, s = i & 7;
    x32[i] = z0[r * ZNUM + s];
  }
  // ---- cache this WG's 128 gate-columns of W in LDS as bf16 (once) ----
  // local tile tt: quadrant q = tt>>1, sub-col cl = tt&1 -> gate col tile q*16 + cg*2 + cl
  for (int idx = tid; idx < 8 * 16 * KDIM; idx += NTHR) {
    int tt  = idx >> 13;            // 0..7
    int col = (idx >> 9) & 15;      // 0..15
    int k   = idx & 511;
    int n   = ((tt >> 1) * 16 + cg * 2 + (tt & 1)) * 16 + col;
    sB[(tt * 16 + col) * LDS_STRIDE + k] = f2bfu(W[(size_t)(l * GDIM + n) * KDIM + k]);
  }
  // ---- alpha/x projection weights in LDS (identical copy per WG) ----
  for (int i = tid; i < SNUM * HNUM; i += NTHR) {
    sAW[i] = aW[i];
    sXW[i] = xWm[i];
  }

  for (int t = 0; t < TNUM; ++t) {
    grid_barrier(bar);   // also covers init-phase + sB fill visibility at t=0

    // ---- stage A slab (64 rows x K=512, bf16) into LDS, vectorized ----
    // h part: k in [256,512), straight bf16 copy, 16B chunks
    for (int i = tid; i < 64 * 32; i += NTHR) {
      int row = i >> 5, c = i & 31;
      int r = r0 + row;
      *(int4*)&sA[row * LDS_STRIDE + 256 + c * 8] =
          *(const int4*)&hbf[(size_t)r * HNUM + c * 8];
    }
    // u part: k in [8,256), fp32->bf16, float4 in / uint2 out
    {
      int row = tid >> 2;
      int b = (r0 + row) & 255;
      const float* urow = u + ((size_t)b * TNUM + t) * DINN;
      for (int c = (tid & 3); c < 62; c += 4) {
        float4 f = *(const float4*)(urow + c * 4);
        uint2 p;
        p.x = (unsigned)f2bfu(f.x) | ((unsigned)f2bfu(f.y) << 16);
        p.y = (unsigned)f2bfu(f.z) | ((unsigned)f2bfu(f.w) << 16);
        *(uint2*)&sA[row * LDS_STRIDE + 8 + c * 4] = p;
      }
    }
    // x part: k in [0,8): broadcast x of last layer
    for (int i = tid; i < 64 * 8; i += NTHR) {
      int row = i >> 3, k = i & 7;
      int b = (r0 + row) & 255;
      sA[row * LDS_STRIDE + k] = f2bfu(x32[(256 + b) * SNUM + k]);
    }
    __syncthreads();

    // ---- WMMA gate GEMM, all operands in LDS; acc[q] = gate quadrant q ----
    {
      const int rtl = wave >> 1;      // local row tile 0..3
      const int clw = wave & 1;       // sub-col 0..1
      v8f acc[4] = {v8f{}, v8f{}, v8f{}, v8f{}};
      const unsigned short* ap =
          sA + (rtl * 16 + (lane & 15)) * LDS_STRIDE + ((lane >> 4) * 8);
      const unsigned short* bp0 = sB + (lane & 15) * LDS_STRIDE + ((lane >> 4) * 8) +
                                  clw * (16 * LDS_STRIDE);

#define LOADFRAGS(AF, BF, K0)                                                  \
  do {                                                                         \
    (AF).q[0] = *(const int4*)(ap + (K0));                                     \
    (AF).q[1] = *(const int4*)(ap + (K0) + 16);                                \
    _Pragma("unroll") for (int q = 0; q < 4; ++q) {                            \
      const unsigned short* bp = bp0 + q * (32 * LDS_STRIDE) + (K0);           \
      (BF)[q].q[0] = *(const int4*)(bp);                                       \
      (BF)[q].q[1] = *(const int4*)(bp + 16);                                  \
    }                                                                          \
  } while (0)

      // Ping-pong double buffer across K-steps: loads for step i+1 are issued
      // into the other register set while step i's 4 WMMAs execute, so the
      // s_wait_dscnt before each WMMA group retires immediately.
      FragBF a0, b0[4], a1, b1[4];
      LOADFRAGS(a0, b0, 0);
#pragma unroll
      for (int k0 = 0; k0 < KDIM; k0 += 64) {
        LOADFRAGS(a1, b1, k0 + 32);
#pragma unroll
        for (int q = 0; q < 4; ++q) {
          acc[q] = __builtin_amdgcn_wmma_f32_16x16x32_bf16(
              false, a0.v, false, b0[q].v, (short)0, acc[q], false, false);
        }
        if (k0 + 64 < KDIM) LOADFRAGS(a0, b0, k0 + 64);
#pragma unroll
        for (int q = 0; q < 4; ++q) {
          acc[q] = __builtin_amdgcn_wmma_f32_16x16x32_bf16(
              false, a1.v, false, b1[q].v, (short)0, acc[q], false, false);
        }
      }
#undef LOADFRAGS

      // ---- fused LSTM cell update straight from the accumulators ----
      const int hcol = (cg * 2 + clw) * 16 + (lane & 15);   // 0..255
      const float bi = bias[l * GDIM + hcol];
      const float bf = bias[l * GDIM + 256 + hcol];
      const float bg = bias[l * GDIM + 512 + hcol];
      const float bo = bias[l * GDIM + 768 + hcol];
#pragma unroll
      for (int rr = 0; rr < 8; ++rr) {
        int r = r0 + rtl * 16 + rr + 8 * (lane >> 4);   // C-layout M index
        int b = r & 255;
        float iv = acc[0][rr] + bi;
        float fv = acc[1][rr] + bf;
        float gv = acc[2][rr] + bg;
        float ov = acc[3][rr] + bo;
        size_t ch = (size_t)r * HNUM + hcol;
        float cn = sigm(fv) * c32[ch] + sigm(iv) * tanhf(gv);
        float hn = sigm(ov) * tanhf(cn);
        c32[ch] = cn;           // WG-private: no cross-WG hazard
        h32[ch] = hn;
        hbf[ch] = f2bfu(hn);
        if (l == 1) out[(size_t)(b * TNUM + t) * ZNUM + SNUM + hcol] = hn;
      }
    }

    grid_barrier(bar);

    // ---- phase 2: one wave per (l,b) row: alpha/x GEMVs + ODE mix ----
    {
      int r = g * 8 + wave;
      int lr = r >> 8, b = r & 255;
      const float* hrow = h32 + (size_t)r * HNUM;
      float da[8] = {0, 0, 0, 0, 0, 0, 0, 0};
      float dx[8] = {0, 0, 0, 0, 0, 0, 0, 0};
#pragma unroll
      for (int jj = 0; jj < 8; ++jj) {
        int j = lane + jj * 32;
        float hn = hrow[j];
#pragma unroll
        for (int s = 0; s < 8; ++s) {
          da[s] += hn * sAW[s * HNUM + j];
          dx[s] += hn * sXW[s * HNUM + j];
        }
      }
#pragma unroll
      for (int s = 0; s < 8; ++s) {
#pragma unroll
        for (int off = 16; off > 0; off >>= 1) {
          da[s] += __shfl_xor(da[s], off, 32);
          dx[s] += __shfl_xor(dx[s], off, 32);
        }
      }
      if (lane == 0) {
        float ta = tau[(size_t)b * TNUM + t];
        float xo[8];
#pragma unroll
        for (int s = 0; s < 8; ++s) xo[s] = x32[r * SNUM + s];
#pragma unroll
        for (int s = 0; s < 8; ++s) {
          float acc = 0.0f;
#pragma unroll
          for (int s2 = 0; s2 < 8; ++s2) acc += Amat[s * SNUM + s2] * xo[s2];
          float xmid = xo[s] + ta * acc;
          float al = sigm(da[s] + ab[s]);
          float xp = dx[s] + xbv[s];
          float xn = al * xmid + (1.0f - al) * xp;
          x32[r * SNUM + s] = xn;
          if (lr == 1) {
            out[(size_t)(b * TNUM + t) * ZNUM + s] = xn;
            out[OFF_COEF + (size_t)(b * TNUM + t) * SNUM + s] = al;
          }
        }
      }
    }
  }

  grid_barrier(bar);

  // ---- epilogue: z_final / c_z_final ----
  for (int i = gt; i < ROWS * ZNUM; i += NWG * NTHR) {
    int r = i / ZNUM, e = i - r * ZNUM;
    float zv, cv;
    if (e < SNUM) {
      zv = x32[r * SNUM + e];
      cv = cz0[r * ZNUM + e];          // cx passthrough
    } else {
      zv = h32[r * HNUM + (e - SNUM)];
      cv = c32[r * HNUM + (e - SNUM)];
    }
    out[OFF_ZF + (size_t)i] = zv;
    out[OFF_CZF + (size_t)i] = cv;
  }
}

extern "C" void kernel_launch(void* const* d_in, const int* in_sizes, int n_in,
                              void* d_out, int out_size, void* d_ws, size_t ws_size,
                              hipStream_t stream) {
  const float* u   = (const float*)d_in[0];
  const float* z0  = (const float*)d_in[1];
  const float* cz0 = (const float*)d_in[2];
  const float* tau = (const float*)d_in[3];
  const float* A   = (const float*)d_in[4];
  const float* W   = (const float*)d_in[5];
  const float* b   = (const float*)d_in[6];
  const float* aW  = (const float*)d_in[7];
  const float* ab  = (const float*)d_in[8];
  const float* xW  = (const float*)d_in[9];
  const float* xb  = (const float*)d_in[10];
  float* out = (float*)d_out;

  char* ws = (char*)d_ws;
  float* x32          = (float*)(ws);                         // 16 KB
  float* c32          = (float*)(ws + (1ull << 20));          // 512 KB
  float* h32          = (float*)(ws + (2ull << 20));          // 512 KB
  unsigned short* hbf = (unsigned short*)(ws + (3ull << 20)); // 256 KB
  unsigned* bar       = (unsigned*)(ws + (4ull << 20));       // 8 B

  bar_init_kernel<<<1, 32, 0, stream>>>(bar);
  ltc_lstm_persistent<<<NWG, NTHR, 0, stream>>>(
      u, z0, cz0, tau, A, W, b, aW, ab, xW, xb,
      out, x32, c32, h32, hbf, bar);
}
